// ConditionInjectionBlock_68135361184177
// MI455X (gfx1250) — matrix operand order repaired and checked by reading, not compiled, not run on hardware
//
#include <hip/hip_runtime.h>
#include <hip/hip_bf16.h>

typedef __attribute__((ext_vector_type(16))) _Float16 v16h;
typedef __attribute__((ext_vector_type(8)))  _Float16 v8h;
typedef __attribute__((ext_vector_type(4)))  _Float16 v4h;
typedef __attribute__((ext_vector_type(8)))  float    v8f;

// Problem constants
#define CDIM   256
#define HEADS  8
#define HD     32
#define NTOK   64     // tokens per 8x8 window
#define NWIN   2048   // 8 * 16 * 16 windows
#define SCALE  0.17677669529663687f  // 32^-0.5

// ---------------------------------------------------------------------------
// WMMA fragment helpers (CDNA5 wave32 layouts, f16 16x16x32)
// A 16x32: lane row m = lane&15; halves [0..7]=K k0..k0+7, [8..15]=K k0+16..k0+23,
//          k0 = (lane>>4)*8 relative to the row+kstep base pointer.
// B 32x16: lane col n = lane&15; halves [0..7]=K k0..k0+7, [8..15]=K k0+16..k0+23 (as A) of the
//          n-major (row = n, K contiguous) base pointer.
// C/D f32 16x16: vgpr r holds (m = r + 8*(lane>>4), n = lane&15).
// ---------------------------------------------------------------------------
__device__ __forceinline__ v16h load_frag_a(const _Float16* base, int hi) {
  v16h r;
  const _Float16* p = base + hi * 8;
  *((v8h*)&r)     = *((const v8h*)p);
  *((v8h*)&r + 1) = *((const v8h*)(p + 16));
  return r;
}

__device__ __forceinline__ v16h load_frag_b(const _Float16* base, int hi) {
  v16h r;
  const _Float16* p = base + hi * 8;
  *((v8h*)&r)     = *((const v8h*)p);
  *((v8h*)&r + 1) = *((const v8h*)(p + 16));
  return r;
}

__device__ __forceinline__ v8f wmma_f16(v16h a, v16h b, v8f c) {
  return __builtin_amdgcn_wmma_f32_16x16x32_f16(false, a, false, b, (short)0, c,
                                                false, false);
}

// ---------------------------------------------------------------------------
// Prep kernels (run once per launch; outputs live in workspace, L2-resident)
// ---------------------------------------------------------------------------

// q = (embedding @ q_w^T + q_b) * SCALE, stored f16 as [head][tok][d] (row len 32)
__global__ void prep_q(const float* __restrict__ emb, const float* __restrict__ qw,
                       const float* __restrict__ qb, _Float16* __restrict__ qf) {
  int o = blockIdx.x * 256 + threadIdx.x;     // 16384 outputs: h*2048 + m*32 + d
  int h = o >> 11;
  int m = (o >> 5) & 63;
  int d = o & 31;
  int drow = h * 32 + d;
  const float* e  = emb + m * CDIM;
  const float* wr = qw + drow * CDIM;
  float s = 0.f;
  for (int c = 0; c < CDIM; ++c) s += e[c] * wr[c];
  qf[o] = (_Float16)((s + qb[drow]) * SCALE);
}

// bias[h][m][j] = rpb_table[rel_idx(m,j)][h]  (f32)
__global__ void prep_bias(const float* __restrict__ rpb, float* __restrict__ biasf) {
  int o = blockIdx.x * 256 + threadIdx.x;     // 32768: h*4096 + m*64 + j
  int h = o >> 12;
  int m = (o >> 6) & 63;
  int j = o & 63;
  int idx = ((m >> 3) - (j >> 3) + 7) * 15 + ((m & 7) - (j & 7) + 7);
  biasf[o] = rpb[idx * HEADS + h];
}

// Convert kv_w (512x256) and proj_w (256x256) to f16 (row-major, K contiguous)
__global__ void prep_w(const float* __restrict__ kvw, const float* __restrict__ pw,
                       _Float16* __restrict__ kvw16, _Float16* __restrict__ pw16) {
  int o = blockIdx.x * 256 + threadIdx.x;     // 196608 total
  if (o < 512 * 256) kvw16[o] = (_Float16)kvw[o];
  else               pw16[o - 512 * 256] = (_Float16)pw[o - 512 * 256];
}

// ---------------------------------------------------------------------------
// Fused per-window kernel: 1 workgroup (8 wave32) per window.
// LDS (dynamic, 147456 B):
//   winls [64][256] f16  (32KB)  -- window tile; reused as attention-out (proj A)
//   kls   [64][256] f16  (32KB)  -- K, [tok][h*32+d]
//   vts   [256][64] f16  (32KB)  -- V transposed, [h*32+d][tok]
//   sscr  8 x [16][64] f32 (32KB) -- per-wave score scratch
//   pscr  8 x [16][64] f16 (16KB) -- per-wave softmax probs
// ---------------------------------------------------------------------------
__global__ void __launch_bounds__(256)
fused_win_attn(const float* __restrict__ x,
               const float* __restrict__ kv_b,
               const float* __restrict__ proj_b,
               const _Float16* __restrict__ qf,     // [8][64][32] f16
               const float* __restrict__ biasf,     // [8][64][64] f32
               const _Float16* __restrict__ kvw16,  // [512][256] f16
               const _Float16* __restrict__ pw16,   // [256][256] f16
               float* __restrict__ out)             // [2048][64][256] f32
{
  extern __shared__ char smem[];
  _Float16* winls = (_Float16*)smem;                 // 16384 halves (also ols)
  _Float16* kls   = winls + NTOK * CDIM;             // 16384 halves
  _Float16* vts   = kls + NTOK * CDIM;               // 16384 halves
  float*    sscr  = (float*)(vts + NTOK * CDIM);     // 8192 floats
  _Float16* pscr  = (_Float16*)(sscr + 8 * 1024);    // 8192 halves

  const int tid  = threadIdx.x;
  const int lane = tid & 31;
  const int w    = tid >> 5;        // wave id 0..7
  const int hi   = lane >> 4;       // lane-half selector
  const int ln   = lane & 15;

  // Window decode: x viewed as (B=8, a=16, p=8, q=16, r=8, C=256); window=(b,a,q)
  const int bw = blockIdx.x;
  const int bb = bw >> 8;
  const int aa = (bw >> 4) & 15;
  const int qq = bw & 15;
  const long xbase = (long)bb * 4194304 + (long)aa * 262144 + (long)qq * 2048;

  // ---- Phase 0: load window tile (f32 -> f16 in LDS, row-major [tok][c]) ----
  {
    const int t  = tid >> 2;                 // token 0..63
    const int c0 = (tid & 3) * 64;           // channel base
    const float* src = x + xbase + (long)(t >> 3) * 32768 + (t & 7) * 256;
    _Float16* dst = winls + t * CDIM;
#pragma unroll
    for (int i = 0; i < 16; ++i) {
      const int c = c0 + i * 4;
      const float4 v = *(const float4*)(src + c);
      v4h h4 = {(_Float16)v.x, (_Float16)v.y, (_Float16)v.z, (_Float16)v.w};
      *(v4h*)(dst + c) = h4;
    }
  }
  __syncthreads();

  // ---- Phase 1: KV GEMM  (64 x 512 x 256); wave w owns n in [w*64, w*64+64) ----
  {
    v8f acc[4][4];
    const v8f vz = {};
#pragma unroll
    for (int mt = 0; mt < 4; ++mt)
#pragma unroll
      for (int nt = 0; nt < 4; ++nt) acc[mt][nt] = vz;

    const _Float16* bbase = kvw16 + (w * 64 + ln) * CDIM;
    for (int ks = 0; ks < 8; ++ks) {
      const int kk = ks * 32;
      v16h a[4], b[4];
#pragma unroll
      for (int mt = 0; mt < 4; ++mt)
        a[mt] = load_frag_a(winls + (mt * 16 + ln) * CDIM + kk, hi);
#pragma unroll
      for (int nt = 0; nt < 4; ++nt)
        b[nt] = load_frag_b(bbase + nt * 16 * CDIM + kk, hi);
#pragma unroll
      for (int mt = 0; mt < 4; ++mt)
#pragma unroll
        for (int nt = 0; nt < 4; ++nt)
          acc[mt][nt] = wmma_f16(a[mt], b[nt], acc[mt][nt]);
    }
    // epilogue: + kv_b, split K/V, f16 into LDS (V transposed)
#pragma unroll
    for (int nt = 0; nt < 4; ++nt) {
      const int n = w * 64 + nt * 16 + ln;
      const float kb = kv_b[n];
      if (n < CDIM) {  // K part (uniform per wave: waves 0..3)
#pragma unroll
        for (int mt = 0; mt < 4; ++mt)
#pragma unroll
          for (int r = 0; r < 8; ++r) {
            const int tok = mt * 16 + r + hi * 8;
            kls[tok * CDIM + n] = (_Float16)(acc[mt][nt][r] + kb);
          }
      } else {         // V part -> transposed [d][tok] (waves 4..7)
#pragma unroll
        for (int mt = 0; mt < 4; ++mt)
#pragma unroll
          for (int r = 0; r < 8; ++r) {
            const int tok = mt * 16 + r + hi * 8;
            vts[(n - CDIM) * NTOK + tok] = (_Float16)(acc[mt][nt][r] + kb);
          }
      }
    }
  }
  __syncthreads();

  // ---- Phase 2: attention; wave w = head w; 4 M-tiles of 16 tokens ----
  {
    const int h = w;
    float* ss = sscr + w * 1024;       // [16][64] f32
    _Float16* ps = pscr + w * 1024;    // [16][64] f16
#pragma unroll 1
    for (int mt = 0; mt < 4; ++mt) {
      // scores: q (A, shared precomputed) x K^T (B = kls rows), K-dim = hd = 32
      const v16h qa = load_frag_a(qf + h * 2048 + (mt * 16 + ln) * HD, hi);
      v8f s[4];
#pragma unroll
      for (int jt = 0; jt < 4; ++jt) {
        const v16h kb = load_frag_b(kls + (jt * 16 + ln) * CDIM + h * HD, hi);
        const v8f z = {};
        s[jt] = wmma_f16(qa, kb, z);
      }
      // spill scores to per-wave scratch [m_loc][j]
#pragma unroll
      for (int jt = 0; jt < 4; ++jt) {
        const int j = jt * 16 + ln;
#pragma unroll
        for (int r = 0; r < 8; ++r)
          ss[(r + hi * 8) * 64 + j] = s[jt][r];
      }
      __syncthreads();

      // softmax over j (64), + rel-pos bias; lanes 0..15 own one row each
      if (lane < 16) {
        const int tok = mt * 16 + lane;
        const float* row  = ss + lane * 64;
        const float* brow = biasf + h * 4096 + tok * 64;
        float vbuf[64];
        float mx = -3.0e38f;
#pragma unroll
        for (int j = 0; j < 64; ++j) {
          const float t = row[j] + brow[j];
          vbuf[j] = t;
          mx = fmaxf(mx, t);
        }
        float sum = 0.f;
#pragma unroll
        for (int j = 0; j < 64; ++j) {
          const float e = __expf(vbuf[j] - mx);
          vbuf[j] = e;
          sum += e;
        }
        const float inv = 1.0f / sum;
        _Float16* prow = ps + lane * 64;
#pragma unroll
        for (int j = 0; j < 64; ++j) prow[j] = (_Float16)(vbuf[j] * inv);
      }
      __syncthreads();

      // out = P (16x64) @ V (64x32): 2 k-steps, 2 n-tiles, B from transposed V
      v8f o[2];
      o[0] = (v8f){};
      o[1] = (v8f){};
#pragma unroll
      for (int ks = 0; ks < 2; ++ks) {
        const v16h pa = load_frag_a(ps + ln * 64 + ks * 32, hi);
#pragma unroll
        for (int ntl = 0; ntl < 2; ++ntl) {
          const v16h vb =
              load_frag_b(vts + (h * HD + ntl * 16 + ln) * NTOK + ks * 32, hi);
          o[ntl] = wmma_f16(pa, vb, o[ntl]);
        }
      }
      // store attention output into winls (reused) as proj A-matrix [tok][c]
#pragma unroll
      for (int ntl = 0; ntl < 2; ++ntl) {
        const int d = h * HD + ntl * 16 + ln;
#pragma unroll
        for (int r = 0; r < 8; ++r) {
          const int tok = mt * 16 + r + hi * 8;
          winls[tok * CDIM + d] = (_Float16)o[ntl][r];
        }
      }
    }
  }
  __syncthreads();

  // ---- Phase 3: proj GEMM (64 x 256 x 256); wave w owns n in [w*32, w*32+32) ----
  {
    v8f acc[4][2];
    const v8f vz = {};
#pragma unroll
    for (int mt = 0; mt < 4; ++mt) { acc[mt][0] = vz; acc[mt][1] = vz; }

    const _Float16* bbase = pw16 + (w * 32 + ln) * CDIM;
    for (int ks = 0; ks < 8; ++ks) {
      const int kk = ks * 32;
      v16h a[4];
#pragma unroll
      for (int mt = 0; mt < 4; ++mt)
        a[mt] = load_frag_a(winls + (mt * 16 + ln) * CDIM + kk, hi);
#pragma unroll
      for (int ntl = 0; ntl < 2; ++ntl) {
        const v16h b = load_frag_b(bbase + ntl * 16 * CDIM + kk, hi);
#pragma unroll
        for (int mt = 0; mt < 4; ++mt)
          acc[mt][ntl] = wmma_f16(a[mt], b, acc[mt][ntl]);
      }
    }
    float* obase = out + (long)bw * (NTOK * CDIM);
#pragma unroll
    for (int ntl = 0; ntl < 2; ++ntl) {
      const int n = w * 32 + ntl * 16 + ln;
      const float pb = proj_b[n];
#pragma unroll
      for (int mt = 0; mt < 4; ++mt)
#pragma unroll
        for (int r = 0; r < 8; ++r) {
          const int tok = mt * 16 + r + hi * 8;
          obase[tok * CDIM + n] = acc[mt][ntl][r] + pb;
        }
    }
  }
}

// ---------------------------------------------------------------------------
extern "C" void kernel_launch(void* const* d_in, const int* in_sizes, int n_in,
                              void* d_out, int out_size, void* d_ws, size_t ws_size,
                              hipStream_t stream) {
  const float* x    = (const float*)d_in[0];
  const float* emb  = (const float*)d_in[1];
  const float* rpb  = (const float*)d_in[2];
  const float* qw   = (const float*)d_in[3];
  const float* qb   = (const float*)d_in[4];
  const float* kvw  = (const float*)d_in[5];
  const float* kvb  = (const float*)d_in[6];
  const float* pw   = (const float*)d_in[7];
  const float* pb   = (const float*)d_in[8];

  char* ws = (char*)d_ws;
  _Float16* qf    = (_Float16*)ws;                  //  32768 B  [8][64][32] f16
  float*    biasf = (float*)(ws + 32768);           // 131072 B  [8][64][64] f32
  _Float16* kvw16 = (_Float16*)(ws + 163840);       // 262144 B  [512][256] f16
  _Float16* pw16  = (_Float16*)(ws + 425984);       // 131072 B  [256][256] f16

  prep_q   <<<64,  256, 0, stream>>>(emb, qw, qb, qf);
  prep_bias<<<128, 256, 0, stream>>>(rpb, biasf);
  prep_w   <<<768, 256, 0, stream>>>(kvw, pw, kvw16, pw16);

  const size_t lds_bytes = (size_t)(NTOK * CDIM * 2) * 3   // winls + kls + vts
                         + (size_t)(8 * 1024) * 4          // sscr
                         + (size_t)(8 * 1024) * 2;         // pscr  = 147456 B
  fused_win_attn<<<NWIN, 256, lds_bytes, stream>>>(x, kvb, pb, qf, biasf,
                                                   kvw16, pw16, (float*)d_out);
}
